// MoireLayer_80169859547624
// MI455X (gfx1250) — compile-verified, matrix operand order checked by reading, not run on hardware
//
#include <hip/hip_runtime.h>
#include <hip/hip_bf16.h>

typedef __attribute__((ext_vector_type(16))) _Float16 v16h;
typedef __attribute__((ext_vector_type(8)))  _Float16 v8h;
typedef __attribute__((ext_vector_type(8)))  float    v8f;

#define BQ   8
#define NQ   1024
#define DQ   512
#define OQ   512
#define HQ   8
#define HDQ  64
#define MROWS (BQ*NQ)   /* 8192 tokens */

// ---------------------------------------------------------------------------
// Fragment load: "A-style" 16x32 f16 fragment from row-major memory.
// Lane L holds row (L&15); elements 0..7 = K (L>>4)*8+0..7, elements 8..15 =
// K 16+(L>>4)*8+0..7  (CDNA5 ISA 7.12.2, 16-bit A-matrix 16x32 layout).
// Works for B operands too whenever memory holds B^T row-major (x@W.T, Q@K.T,
// P@V with V stored transposed).
// ---------------------------------------------------------------------------
__device__ __forceinline__ v16h load_a16(const _Float16* __restrict__ base,
                                         int stride, int lane) {
  const int row = lane & 15;
  const int co  = (lane >> 4) << 3;
  const _Float16* p = base + row * stride + co;
  v8h lo = *(const v8h*)(p);
  v8h hi = *(const v8h*)(p + 16);
  v16h r;
#pragma unroll
  for (int i = 0; i < 8; ++i) { r[i] = lo[i]; r[i + 8] = hi[i]; }
  return r;
}

__device__ __forceinline__ v8f vzero8() {
  v8f z;
#pragma unroll
  for (int i = 0; i < 8; ++i) z[i] = 0.0f;
  return z;
}

__device__ __forceinline__ v8f wmma16(v16h a, v16h b, v8f c) {
  return __builtin_amdgcn_wmma_f32_16x16x32_f16(false, a, false, b,
                                                (short)0, c, false, false);
}

// ---------------------------------------------------------------------------
// Elementwise f32 -> f16 conversion
// ---------------------------------------------------------------------------
__global__ void __launch_bounds__(256)
k_f32_to_f16(const float* __restrict__ src, _Float16* __restrict__ dst, int n) {
  int i = blockIdx.x * blockDim.x + threadIdx.x;
  int stride = gridDim.x * blockDim.x;
  for (; i < n; i += stride) dst[i] = (_Float16)src[i];
}

// ---------------------------------------------------------------------------
// 64x64 wave tile core: D += A[64xK] * B[Kx64], B supplied as 4 row-major
// (B^T) sub-tile base pointers (16 output columns each).
// 16 v8f accumulators (128 VGPRs) + 8 live fragments (64 VGPRs): needs the
// enlarged VGPR budget from __launch_bounds__(128,1) on the callers so the
// accumulators never spill to scratch.
// ---------------------------------------------------------------------------
__device__ __forceinline__ void gemm_core_64x64(
    const _Float16* __restrict__ Abase, int lda,
    const _Float16* const* __restrict__ Bp, int ldb,
    int K, int lane, v8f acc[4][4]) {
#pragma unroll
  for (int im = 0; im < 4; ++im)
#pragma unroll
    for (int in = 0; in < 4; ++in) acc[im][in] = vzero8();

  for (int k0 = 0; k0 < K; k0 += 32) {
    v16h a[4], b[4];
#pragma unroll
    for (int im = 0; im < 4; ++im)
      a[im] = load_a16(Abase + (im * 16) * lda + k0, lda, lane);
#pragma unroll
    for (int in = 0; in < 4; ++in)
      b[in] = load_a16(Bp[in] + k0, ldb, lane);
#pragma unroll
    for (int im = 0; im < 4; ++im)
#pragma unroll
      for (int in = 0; in < 4; ++in)
        acc[im][in] = wmma16(a[im], b[in], acc[im][in]);
  }
}

// ---------------------------------------------------------------------------
// Fused QKV + residual projection:
//   Y[8192, 2048] = x_f16 @ [qkv_w ; wr].T    (K = 512)
// Epilogue scatters into Qf/Kf [b,h,n,hd] (f16), Vt [b,h,hd,n] (f16, transposed
// for the P@V WMMA), xproj [m, 512] (f32).
// Block = 128 threads = 4 waves, each wave one 64x64 tile.
// ---------------------------------------------------------------------------
__global__ void __launch_bounds__(128, 1)
k_gemm_qkv(const _Float16* __restrict__ xf,
           const _Float16* __restrict__ qkvwf,
           const _Float16* __restrict__ wrf,
           const float* __restrict__ qkv_b,
           const float* __restrict__ br,
           _Float16* __restrict__ Qf,
           _Float16* __restrict__ Kf,
           _Float16* __restrict__ Vt,
           float* __restrict__ xproj) {
  const int lane = threadIdx.x & 31;
  const int wid  = blockIdx.x * (blockDim.x >> 5) + (threadIdx.x >> 5);
  const int ntN  = (3 * OQ + OQ) / 64;          // 32 column tiles
  const int m0   = (wid / ntN) * 64;
  const int n0   = (wid % ntN) * 64;

  const _Float16* Bp[4];
#pragma unroll
  for (int in = 0; in < 4; ++in) {
    int c0 = n0 + in * 16;
    Bp[in] = (c0 < 3 * OQ) ? (qkvwf + (size_t)c0 * DQ)
                           : (wrf + (size_t)(c0 - 3 * OQ) * DQ);
  }

  v8f acc[4][4];
  gemm_core_64x64(xf + (size_t)m0 * DQ, DQ, Bp, DQ, DQ, lane, acc);

#pragma unroll
  for (int im = 0; im < 4; ++im) {
#pragma unroll
    for (int r = 0; r < 8; ++r) {
      const int m = m0 + im * 16 + r + ((lane >> 4) << 3);
      const int b = m >> 10;          // batch
      const int n = m & 1023;         // token
#pragma unroll
      for (int in = 0; in < 4; ++in) {
        const int c = n0 + in * 16 + (lane & 15);
        float v = acc[im][in][r];
        if (c < 3 * OQ) {
          v += qkv_b[c];
          const int s  = c >> 9;             // 0=Q 1=K 2=V
          const int hh = (c >> 6) & (HQ - 1);
          const int d  = c & (HDQ - 1);
          const int bh = b * HQ + hh;
          if (s == 0)
            Qf[((size_t)bh * NQ + n) * HDQ + d] = (_Float16)v;
          else if (s == 1)
            Kf[((size_t)bh * NQ + n) * HDQ + d] = (_Float16)v;
          else
            Vt[((size_t)bh * HDQ + d) * NQ + n] = (_Float16)v;
        } else {
          const int cc = c - 3 * OQ;
          xproj[(size_t)m * OQ + cc] = v + br[cc];
        }
      }
    }
  }
}

// ---------------------------------------------------------------------------
// Attention: one workgroup (2 waves, 64 threads) per (b,h, 32-row block).
// Wave w owns 16 rows. Scores staged in 128KB LDS (fits 320KB/WGP),
// two-pass softmax, P@V via WMMA, row-masked f16 output.
// ---------------------------------------------------------------------------
__global__ void __launch_bounds__(64, 1)
k_attn(const _Float16* __restrict__ Qf,
       const _Float16* __restrict__ Kf,
       const _Float16* __restrict__ Vt,
       const float* __restrict__ slw,
       const int* __restrict__ mask,
       _Float16* __restrict__ hm) {
  extern __shared__ float smem[];
  float* sc   = smem;                 // [32][1024] scores
  float* pmax = smem + 32 * NQ;       // [64]
  float* rmax = pmax + 64;            // [32]
  float* psum = rmax + 32;            // [64]
  float* rsum = psum + 64;            // [32]

  const int lane = threadIdx.x & 31;
  const int w    = threadIdx.x >> 5;  // wave 0..1
  const int nblk = NQ / 32;
  const int bh   = blockIdx.x / nblk;
  const int rb   = blockIdx.x % nblk;
  const int b    = bh >> 3;
  const int h    = bh & (HQ - 1);
  const int rowstart = rb * 32 + w * 16;   // this wave's first global row

  // Q fragments for this wave's 16 rows (K = 64 -> two 32-chunks), loaded once
  const _Float16* Qbase = Qf + ((size_t)bh * NQ + rowstart) * HDQ;
  const v16h aq0 = load_a16(Qbase,      HDQ, lane);
  const v16h aq1 = load_a16(Qbase + 32, HDQ, lane);

  const float slwh  = slw[h];
  const float scale = 0.125f;  // 1/sqrt(64)

  // ---- scores: S = Q K^T / 8 + diag bias, column mask, stage to LDS ----
  for (int n0 = 0; n0 < NQ; n0 += 16) {
    const _Float16* Kbase = Kf + ((size_t)bh * NQ + n0) * HDQ;
    v16h bk0 = load_a16(Kbase,      HDQ, lane);
    v16h bk1 = load_a16(Kbase + 32, HDQ, lane);
    v8f s = vzero8();
    s = wmma16(aq0, bk0, s);
    s = wmma16(aq1, bk1, s);

    const int  col   = n0 + (lane & 15);
    const bool colok = (mask[b * NQ + col] != 0);
#pragma unroll
    for (int r = 0; r < 8; ++r) {
      const int mrow = rowstart + r + ((lane >> 4) << 3);
      float v = s[r] * scale;
      if (mrow == col) v += slwh;        // eye * self_loop_W[h]
      v = colok ? v : -1.0e6f;           // padding mask (columns)
      const int lrow = w * 16 + r + ((lane >> 4) << 3);
      sc[lrow * NQ + col] = v;
    }
  }
  __syncthreads();

  // ---- softmax over 1024 cols: 2 threads per row ----
  {
    const int t    = threadIdx.x;
    const int row  = t >> 1;
    const int half = t & 1;
    float* prow = sc + row * NQ + half * 512;
    float mx = -3.0e38f;
    for (int i = 0; i < 512; ++i) mx = fmaxf(mx, prow[i]);
    pmax[t] = mx;
    __syncthreads();
    if (half == 0) rmax[row] = fmaxf(pmax[2 * row], pmax[2 * row + 1]);
    __syncthreads();
    const float rm = rmax[row];
    float sum = 0.0f;
    for (int i = 0; i < 512; ++i) {
      float e = __expf(prow[i] - rm);
      prow[i] = e;
      sum += e;
    }
    psum[t] = sum;
    __syncthreads();
    if (half == 0) rsum[row] = psum[2 * row] + psum[2 * row + 1];
    __syncthreads();
  }

  // ---- P @ V: out[16 x 64] per wave ----
  v8f accd[4];
#pragma unroll
  for (int dt = 0; dt < 4; ++dt) accd[dt] = vzero8();

  const int co   = (lane >> 4) << 3;
  const int arow = w * 16 + (lane & 15);
  for (int n0 = 0; n0 < NQ; n0 += 32) {
    const float* prow = sc + arow * NQ + n0 + co;
    v16h a;
#pragma unroll
    for (int i = 0; i < 8; ++i) {
      a[i]     = (_Float16)prow[i];
      a[i + 8] = (_Float16)prow[i + 16];
    }
#pragma unroll
    for (int dt = 0; dt < 4; ++dt) {
      const _Float16* Vb = Vt + ((size_t)bh * HDQ + dt * 16) * NQ + n0;
      v16h bv = load_a16(Vb, NQ, lane);
      accd[dt] = wmma16(a, bv, accd[dt]);
    }
  }

  // normalize by row-sum, apply row mask, write f16 for FFN input
#pragma unroll
  for (int r = 0; r < 8; ++r) {
    const int lrow = w * 16 + r + ((lane >> 4) << 3);
    const int grow = rb * 32 + lrow;
    const float fm  = (mask[b * NQ + grow] != 0) ? 1.0f : 0.0f;
    const float f   = fm / rsum[lrow];
#pragma unroll
    for (int dt = 0; dt < 4; ++dt) {
      const int d = dt * 16 + (lane & 15);
      hm[((size_t)(b * NQ + grow)) * OQ + h * HDQ + d] =
          (_Float16)(accd[dt][r] * f);
    }
  }
}

// ---------------------------------------------------------------------------
// FFN GEMMs (512x512 weights, K=512).
// MODE 0: t = relu(h @ w1.T + b1) -> f16
// MODE 1: out = 0.5*fmask*(t @ w2.T + b2) + 0.5*xproj -> f32
// ---------------------------------------------------------------------------
template <int MODE>
__global__ void __launch_bounds__(128, 1)
k_gemm_ffn(const _Float16* __restrict__ Af,
           const _Float16* __restrict__ Wf,
           const float* __restrict__ bias,
           const int* __restrict__ mask,
           const float* __restrict__ xproj,
           _Float16* __restrict__ outh,
           float* __restrict__ outf) {
  const int lane = threadIdx.x & 31;
  const int wid  = blockIdx.x * (blockDim.x >> 5) + (threadIdx.x >> 5);
  const int ntN  = OQ / 64;   // 8
  const int m0   = (wid / ntN) * 64;
  const int n0   = (wid % ntN) * 64;

  const _Float16* Bp[4];
#pragma unroll
  for (int in = 0; in < 4; ++in)
    Bp[in] = Wf + (size_t)(n0 + in * 16) * OQ;

  v8f acc[4][4];
  gemm_core_64x64(Af + (size_t)m0 * OQ, OQ, Bp, OQ, OQ, lane, acc);

#pragma unroll
  for (int im = 0; im < 4; ++im) {
#pragma unroll
    for (int r = 0; r < 8; ++r) {
      const int m = m0 + im * 16 + r + ((lane >> 4) << 3);
#pragma unroll
      for (int in = 0; in < 4; ++in) {
        const int c = n0 + in * 16 + (lane & 15);
        float v = acc[im][in][r] + bias[c];
        if (MODE == 0) {
          v = fmaxf(v, 0.0f);
          outh[(size_t)m * OQ + c] = (_Float16)v;
        } else {
          const float fm = (mask[m] != 0) ? 1.0f : 0.0f;
          outf[(size_t)m * OQ + c] =
              0.5f * fm * v + 0.5f * xproj[(size_t)m * OQ + c];
        }
      }
    }
  }
}

// ---------------------------------------------------------------------------
extern "C" void kernel_launch(void* const* d_in, const int* in_sizes, int n_in,
                              void* d_out, int out_size, void* d_ws,
                              size_t ws_size, hipStream_t stream) {
  const float* x    = (const float*)d_in[0];
  /* d_in[1] = adj : unused by reference math */
  const int*   mask = (const int*)d_in[2];
  const float* slw  = (const float*)d_in[3];
  const float* qkvw = (const float*)d_in[4];
  const float* qkvb = (const float*)d_in[5];
  const float* w1   = (const float*)d_in[6];
  const float* b1   = (const float*)d_in[7];
  const float* w2   = (const float*)d_in[8];
  const float* b2   = (const float*)d_in[9];
  const float* wr   = (const float*)d_in[10];
  const float* br   = (const float*)d_in[11];
  float* out = (float*)d_out;

  char* p = (char*)d_ws;
  _Float16* xf    = (_Float16*)p; p += (size_t)MROWS * DQ * 2;
  _Float16* qkvwf = (_Float16*)p; p += (size_t)3 * OQ * DQ * 2;
  _Float16* wrf   = (_Float16*)p; p += (size_t)OQ * DQ * 2;
  _Float16* w1f   = (_Float16*)p; p += (size_t)OQ * OQ * 2;
  _Float16* w2f   = (_Float16*)p; p += (size_t)OQ * OQ * 2;
  _Float16* Qf    = (_Float16*)p; p += (size_t)BQ * HQ * NQ * HDQ * 2;
  _Float16* Kfp   = (_Float16*)p; p += (size_t)BQ * HQ * NQ * HDQ * 2;
  _Float16* Vt    = (_Float16*)p; p += (size_t)BQ * HQ * NQ * HDQ * 2;
  float*    xproj = (float*)p;    p += (size_t)MROWS * OQ * 4;
  _Float16* hm    = (_Float16*)p; p += (size_t)MROWS * OQ * 2;
  _Float16* tf    = (_Float16*)p; p += (size_t)MROWS * OQ * 2;

  // 1) f32 -> f16 operand conversion
  k_f32_to_f16<<<512, 256, 0, stream>>>(x, xf, MROWS * DQ);
  k_f32_to_f16<<<256, 256, 0, stream>>>(qkvw, qkvwf, 3 * OQ * DQ);
  k_f32_to_f16<<<128, 256, 0, stream>>>(wr, wrf, OQ * DQ);
  k_f32_to_f16<<<128, 256, 0, stream>>>(w1, w1f, OQ * OQ);
  k_f32_to_f16<<<128, 256, 0, stream>>>(w2, w2f, OQ * OQ);

  // 2) fused QKV + residual projection: 128 x 32 tiles, 4 waves/block
  k_gemm_qkv<<<(128 * 32) / 4, 128, 0, stream>>>(xf, qkvwf, wrf, qkvb, br,
                                                 Qf, Kfp, Vt, xproj);

  // 3) attention: (b,h) x 32-row blocks, 2 waves, 128KB+ dynamic LDS
  const size_t shmem = (size_t)(32 * NQ + 64 + 32 + 64 + 32) * sizeof(float);
  k_attn<<<BQ * HQ * (NQ / 32), 64, shmem, stream>>>(Qf, Kfp, Vt, slw, mask, hm);

  // 4) FFN
  k_gemm_ffn<0><<<(128 * 8) / 4, 128, 0, stream>>>(hm, w1f, b1, nullptr,
                                                   nullptr, tf, nullptr);
  k_gemm_ffn<1><<<(128 * 8) / 4, 128, 0, stream>>>(tf, w2f, b2, mask, xproj,
                                                   nullptr, out);
}
